// PredRNNv2_64527588655227
// MI455X (gfx1250) — matrix operand order repaired and checked by reading, not compiled, not run on hardware
//
#include <hip/hip_runtime.h>
#include <cmath>

typedef __attribute__((ext_vector_type(16))) _Float16 v16h;
typedef __attribute__((ext_vector_type(8)))  _Float16 h8;
typedef __attribute__((ext_vector_type(4)))  _Float16 h4;
typedef __attribute__((ext_vector_type(8)))  float    v8f;

#define HW      64
#define PIX     4096
#define BATCH   8
#define TSTEPS  12
#define HORIZON 12

// LDS layout (dynamic): As[4][66][64] f16 tile, then Bs[KS*KS][64][32] f16 weights
#define SMEM_A_ELEMS (4 * 66 * 64)

// ---------------------------------------------------------------------------
// Generic implicit-GEMM conv (3x3 pad1 or 1x1 pad0) via v_wmma_f32_16x16x32_f16
// Input/output NHWC. Workgroup: 128 pixels (2 rows x 64 cols); loops over
// `nblocks` 64-wide output-channel blocks reusing the staged activation tile.
// Each wave32 owns a 32x32 patch = 2x2 WMMA tiles.
// ---------------------------------------------------------------------------
struct ConvSeg {
    const float* in;         // input base (time-sliced already)
    const float* w;          // OIHW weight base for this segment
    long long    in_bstride; // floats per batch item
    int          C;          // real input channels
    int          choff;      // output channel offset of block 0
    int          nblocks;    // number of 64-wide output blocks
};
struct ConvArgs {
    float* out;
    int    OCs;              // output channel stride (NHWC)
    ConvSeg seg[3];
};

template <int KS>
__global__ __launch_bounds__(256) void conv_wmma_kernel(ConvArgs cargs)
{
    constexpr int KS2 = KS * KS;
    const ConvSeg s = cargs.seg[blockIdx.y];
    const int tid  = threadIdx.x;
    const int lane = tid & 31;
    const int wave = tid >> 5;
    const int half = lane >> 4;      // lane group (0: lanes0-15, 1: lanes16-31)
    const int mrow = lane & 15;
    const int wm   = wave & 3;       // 4 waves along M
    const int wn   = wave >> 2;      // 2 waves along N
    const int rowInTile = wm >> 1;   // 0..1
    const int colbase   = (wm & 1) * 32;
    const int b  = blockIdx.x >> 5;
    const int y0 = (blockIdx.x & 31) * 2;

    extern __shared__ _Float16 smem[];
    _Float16* As = smem;                  // [row4][col66][ch64]
    _Float16* Bs = smem + SMEM_A_ELEMS;   // [tap][outch64][inch32]

    const int C   = s.C;
    const int nco = (C + 63) >> 6;
    const float* inB = s.in + (long long)b * s.in_bstride;

    // per-lane invariant LDS fragment bases
    int abase0[2];
    #pragma unroll
    for (int mi = 0; mi < 2; ++mi)
        abase0[mi] = (rowInTile * 66 + colbase + 16 * mi + mrow) * 64 + 8 * half;
    int bbase0[2];
    #pragma unroll
    for (int ni = 0; ni < 2; ++ni)
        bbase0[ni] = (wn * 32 + 16 * ni + mrow) * 32 + 8 * half;

    const int gy = y0 + rowInTile;

    for (int nb = 0; nb < s.nblocks; ++nb) {
        v8f acc[2][2];
        #pragma unroll
        for (int mi = 0; mi < 2; ++mi)
            #pragma unroll
            for (int ni = 0; ni < 2; ++ni) { v8f z = {}; acc[mi][ni] = z; }

        const float* wseg = s.w + (long long)nb * 64 * C * KS2;

        for (int co = 0; co < nco; ++co) {
            const int c0 = co * 64;
            const int Cc = (C - c0 < 64) ? (C - c0) : 64;

            // ---- stage activation tile once per co (reused across nb when nco==1)
            if (nb == 0 || nco > 1) {
                __syncthreads();         // prior reads of As done
                for (int pair = tid; pair < 4 * 66; pair += 256) {
                    int r   = pair / 66;
                    int c66 = pair - r * 66;
                    int gyy = y0 + r - 1;
                    int gx  = c66 - 1;
                    bool valid = (gyy >= 0) & (gyy < HW) & (gx >= 0) & (gx < HW);
                    _Float16* dst = &As[pair * 64];
                    if (valid) {
                        const float* src = inB + (long long)(gyy * HW + gx) * C + c0;
                        int c = 0;
                        if ((C & 3) == 0) {
                            for (; c + 4 <= Cc; c += 4) {
                                float4 f = *(const float4*)(src + c);
                                h4 hv;
                                hv.x = (_Float16)f.x; hv.y = (_Float16)f.y;
                                hv.z = (_Float16)f.z; hv.w = (_Float16)f.w;
                                *(h4*)(dst + c) = hv;
                            }
                        }
                        for (; c < Cc; ++c) dst[c] = (_Float16)src[c];
                        for (int cz = Cc; cz < 64; ++cz) dst[cz] = (_Float16)0.f;
                    } else {
                        for (int c = 0; c < 64; c += 4) { h4 z = {}; *(h4*)(dst + c) = z; }
                    }
                }
            }

            const int kch = (Cc + 31) >> 5;
            for (int kc = 0; kc < kch; ++kc) {
                __syncthreads();          // prior compute's As/Bs reads done
                // ---- stage ALL taps of this 32-channel weight chunk ----
                // idx order: tap fastest -> global reads contiguous along (cc,tap)
                {
                    const int cg0 = c0 + kc * 32;
                    for (int idx = tid; idx < KS2 * 2048; idx += 256) {
                        int n   = idx / (32 * KS2);
                        int rem = idx - n * (32 * KS2);
                        int cc  = rem / KS2;
                        int tap = rem - cc * KS2;
                        int gc  = cg0 + cc;
                        float wv = (gc < C)
                                 ? wseg[((long long)n * C + gc) * KS2 + tap]
                                 : 0.f;
                        Bs[tap * 2048 + n * 32 + cc] = (_Float16)wv;
                    }
                }
                __syncthreads();

                const int kcc = kc * 32;
                #pragma unroll
                for (int tap = 0; tap < KS2; ++tap) {
                    const int dy = (KS == 3) ? (tap / 3) : 1;
                    const int dx = (KS == 3) ? (tap - dy * 3) : 1;
                    const int aoff = (dy * 66 + dx) * 64 + kcc;

                    // A frags: lane row = mrow; K(e)=8*half+e (e<8), +16 for e>=8
                    v16h afr[2];
                    #pragma unroll
                    for (int mi = 0; mi < 2; ++mi) {
                        int base = abase0[mi] + aoff;
                        h8 lo = *(const h8*)&As[base];
                        h8 hi = *(const h8*)&As[base + 16];
                        afr[mi] = __builtin_shufflevector(lo, hi,
                            0, 1, 2, 3, 4, 5, 6, 7, 8, 9, 10, 11, 12, 13, 14, 15);
                    }
                    // B frags: lane col = mrow; K(e)=8*half+e (e<8), +16 for e>=8
                    v16h bfr[2];
                    #pragma unroll
                    for (int ni = 0; ni < 2; ++ni) {
                        h8 blo = *(const h8*)&Bs[bbase0[ni] + tap * 2048];
                        h8 bhi = *(const h8*)&Bs[bbase0[ni] + tap * 2048 + 16];
                        bfr[ni] = __builtin_shufflevector(blo, bhi,
                            0, 1, 2, 3, 4, 5, 6, 7, 8, 9, 10, 11, 12, 13, 14, 15);
                    }

                    #pragma unroll
                    for (int mi = 0; mi < 2; ++mi)
                        #pragma unroll
                        for (int ni = 0; ni < 2; ++ni)
                            acc[mi][ni] = __builtin_amdgcn_wmma_f32_16x16x32_f16(
                                false, afr[mi], false, bfr[ni],
                                (short)0, acc[mi][ni], false, false);
                }
            }
        }

        // ---- store block nb: D layout lane = N, VGPR v -> M = v + 8*half ----
        #pragma unroll
        for (int mi = 0; mi < 2; ++mi) {
            #pragma unroll
            for (int ni = 0; ni < 2; ++ni) {
                int ch = s.choff + nb * 64 + wn * 32 + ni * 16 + mrow;
                float tmp[8];
                #pragma unroll
                for (int v = 0; v < 8; ++v) tmp[v] = acc[mi][ni][v];
                #pragma unroll
                for (int v = 0; v < 8; ++v) {
                    int col = colbase + 16 * mi + v + 8 * half;
                    long long o = ((long long)b * PIX + gy * HW + col) * cargs.OCs + ch;
                    cargs.out[o] = tmp[v];
                }
            }
        }
    }
}

// ---------------------------------------------------------------------------
// Pointwise kernels
// ---------------------------------------------------------------------------
__device__ __forceinline__ float sigm(float x) { return 1.f / (1.f + expf(-x)); }

__global__ void gates1_kernel(const float* __restrict__ G,
                              const float* __restrict__ bx,
                              const float* __restrict__ bh,
                              const float* __restrict__ bm,
                              float* cbuf, const float* m_in, float* m_out,
                              float* mem, float* oacc, int total)
{
    int i = blockIdx.x * blockDim.x + threadIdx.x;
    if (i >= total) return;
    int n = i & 63;
    long long bp = (long long)(i >> 6);
    const float* g = G + bp * 896;
    float ix  = g[n]       + bx[n];
    float fx  = g[64 + n]  + bx[64 + n];
    float gx  = g[128 + n] + bx[128 + n];
    float ixp = g[192 + n] + bx[192 + n];
    float fxp = g[256 + n] + bx[256 + n];
    float gxp = g[320 + n] + bx[320 + n];
    float ox  = g[384 + n] + bx[384 + n];
    float ih  = g[448 + n] + bh[n];
    float fh  = g[512 + n] + bh[64 + n];
    float gh  = g[576 + n] + bh[128 + n];
    float oh  = g[640 + n] + bh[192 + n];
    float im  = g[704 + n] + bm[n];
    float fm  = g[768 + n] + bm[64 + n];
    float gm  = g[832 + n] + bm[128 + n];

    float c_old = cbuf[i];
    float m_old = m_in[i];
    float it = sigm(ix + ih);
    float ft = sigm(fx + fh + 1.f);
    float gt = tanhf(gx + gh);
    float cn = ft * c_old + it * gt;
    float itp = sigm(ixp + im);
    float ftp = sigm(fxp + fm + 1.f);
    float gtp = tanhf(gxp + gm);
    float mn = ftp * m_old + itp * gtp;

    cbuf[i]  = cn;
    m_out[i] = mn;
    mem[bp * 128 + n]      = cn;
    mem[bp * 128 + 64 + n] = mn;
    oacc[i] = ox + oh;
}

__global__ void gates2_kernel(const float* __restrict__ oacc,
                              const float* __restrict__ ol,
                              const float* __restrict__ bo,
                              const float* __restrict__ bl,
                              float* hbuf, int total)
{
    int i = blockIdx.x * blockDim.x + threadIdx.x;
    if (i >= total) return;
    int n = i & 63;
    long long bp = (long long)(i >> 6);
    float o = sigm(oacc[i] + ol[bp * 128 + n] + bo[n]);
    hbuf[i] = o * tanhf(ol[bp * 128 + 64 + n] + bl[n]);
}

struct PrepArgs {
    const float* x;
    const float* ls;
    const float* w1[4];  // u,v,t,f
    const float* w2[4];
    const float* b1[4];
    const float* b2[4];
    float* xn;
};

__global__ void prep_kernel(PrepArgs pa)
{
    int i = blockIdx.x * blockDim.x + threadIdx.x;
    if (i >= BATCH * TSTEPS * PIX) return;
    int p  = i & (PIX - 1);
    int bt = i >> 12;
    float lsr = pa.ls[bt] * 0.017453292519943295f;
    const float* xb = pa.x + (long long)bt * 5 * PIX;
    float* o = pa.xn + (long long)i * 9;
    o[0] = xb[p];
    #pragma unroll
    for (int f = 0; f < 4; ++f) {
        float ch = xb[(f + 1) * PIX + p];
        o[1 + 2 * f] = ch * (pa.w1[f][p] * sinf(lsr + pa.b1[f][p]));
        o[2 + 2 * f] = ch * (pa.w2[f][p] * cosf(lsr + pa.b2[f][p]));
    }
}

__global__ void pred_kernel(const float* __restrict__ h,
                            const float* __restrict__ wl,
                            const float* __restrict__ bl,
                            float* out, int t)
{
    int i = blockIdx.x * blockDim.x + threadIdx.x;
    if (i >= BATCH * PIX) return;
    int p = i & (PIX - 1);
    int b = i >> 12;
    const float* hp = h + (long long)i * 64;
    float s = bl[0];
    #pragma unroll
    for (int c = 0; c < 64; ++c) s += hp[c] * wl[c];
    out[((long long)b * HORIZON + t) * PIX + p] = s;
}

// ---------------------------------------------------------------------------
// Host orchestration
// ---------------------------------------------------------------------------
struct CellP { const float *wx,*bx,*wh,*bh,*wm,*bm,*wo,*bo,*wl,*bl; };
struct NetP {
    const float *x, *ls;
    CellP cell[3];
    const float *w_last, *b_last;
    const float *w1[4], *w2[4], *b1[4], *b2[4]; // order u,v,t,f
};

extern "C" void kernel_launch(void* const* d_in, const int* in_sizes, int n_in,
                              void* d_out, int out_size, void* d_ws, size_t ws_size,
                              hipStream_t stream)
{
    if (n_in < 50) return;
    auto F = [&](int i) { return (const float*)d_in[i]; };

    NetP P{};
    // sorted-key param layout helper (jax tree_flatten sorts dict keys):
    // b1_{f,t,u,v}, b2_{f,t,u,v}, b_last, cells[3]{bh,bl,bm,bo,bx,wh,wl,wm,wo,wx},
    // w1_{f,t,u,v}, w2_{f,t,u,v}, w_last
    auto fillSorted = [&](int base) {
        static const int mp[4] = {3, 2, 0, 1}; // sorted(f,t,u,v) -> our(u,v,t,f)
        for (int j = 0; j < 4; ++j) P.b1[mp[j]] = F(base + j);
        for (int j = 0; j < 4; ++j) P.b2[mp[j]] = F(base + 4 + j);
        P.b_last = F(base + 8);
        for (int li = 0; li < 3; ++li) {
            int cb = base + 9 + 10 * li;
            CellP& c = P.cell[li];
            c.bh = F(cb);     c.bl = F(cb + 1); c.bm = F(cb + 2);
            c.bo = F(cb + 3); c.bx = F(cb + 4); c.wh = F(cb + 5);
            c.wl = F(cb + 6); c.wm = F(cb + 7); c.wo = F(cb + 8);
            c.wx = F(cb + 9);
        }
        for (int j = 0; j < 4; ++j) P.w1[mp[j]] = F(base + 39 + j);
        for (int j = 0; j < 4; ++j) P.w2[mp[j]] = F(base + 43 + j);
        P.w_last = F(base + 47);
    };

    if (in_sizes[0] == 96) {                       // mode B: ls, params(sorted), x
        P.ls = F(0);
        fillSorted(1);
        P.x = F(n_in - 1);
    } else if (in_sizes[0] == 1966080 && in_sizes[2] == 4096) { // x, ls, params(sorted)
        P.x = F(0); P.ls = F(1);
        fillSorted(2);
    } else {                                       // mode A: dict insertion order
        P.x = F(0); P.ls = F(1);
        for (int li = 0; li < 3; ++li) {
            int cb = 2 + 10 * li;
            CellP& c = P.cell[li];
            c.wx = F(cb);     c.bx = F(cb + 1); c.wh = F(cb + 2);
            c.bh = F(cb + 3); c.wm = F(cb + 4); c.bm = F(cb + 5);
            c.wo = F(cb + 6); c.bo = F(cb + 7); c.wl = F(cb + 8);
            c.bl = F(cb + 9);
        }
        P.w_last = F(32); P.b_last = F(33);
        int k = 34;
        for (int f = 0; f < 4; ++f) {
            P.w1[f] = F(k++); P.w2[f] = F(k++);
            P.b1[f] = F(k++); P.b2[f] = F(k++);
        }
    }

    // ---- workspace carve (floats) ----
    float* ws = (float*)d_ws;
    size_t off = 0;
    auto alloc = [&](size_t n) { float* p = ws + off; off += (n + 15) & ~(size_t)15; return p; };
    const size_t T64 = (size_t)BATCH * PIX * 64;
    float* xnew = alloc((size_t)BATCH * TSTEPS * PIX * 9);
    float* hb[3], *cb[3];
    float* state0 = nullptr;
    for (int i = 0; i < 3; ++i) { hb[i] = alloc(T64); if (!state0) state0 = hb[i]; }
    for (int i = 0; i < 3; ++i) cb[i] = alloc(T64);
    float* m0    = alloc(T64);
    float* m1    = alloc(T64);
    float* gates = alloc((size_t)BATCH * PIX * 896);
    float* mem   = alloc((size_t)BATCH * PIX * 128);
    float* oacc  = alloc(T64);
    float* ol    = alloc((size_t)BATCH * PIX * 128);
    if (off * sizeof(float) > ws_size) return;

    // zero h[3], c[3], m0, m1 (contiguous)
    hipMemsetAsync(state0, 0, 8 * T64 * sizeof(float), stream);

    // preprocess modulated inputs -> NHWC x_new [B][T][H][W][9]
    {
        PrepArgs pa;
        pa.x = P.x; pa.ls = P.ls; pa.xn = xnew;
        for (int f = 0; f < 4; ++f) {
            pa.w1[f] = P.w1[f]; pa.w2[f] = P.w2[f];
            pa.b1[f] = P.b1[f]; pa.b2[f] = P.b2[f];
        }
        int tot = BATCH * TSTEPS * PIX;
        prep_kernel<<<(tot + 255) / 256, 256, 0, stream>>>(pa);
    }

    const int totalG = (int)T64;
    const dim3 pwGrid((totalG + 255) / 256);
    const size_t smem3 = (SMEM_A_ELEMS + 9 * 64 * 32) * sizeof(_Float16); // 70656 B
    const size_t smem1 = (SMEM_A_ELEMS + 1 * 64 * 32) * sizeof(_Float16); // 37888 B

    float* mc_ = m0;
    float* mn_ = m1;

    auto cell_step = [&](int li, const float* cur, long long cur_bs, int Cin) {
        const CellP& c = P.cell[li];
        // --- fused gate GEMMs: 3 segments (xc x7, hc x4, mc x3 output blocks) ---
        ConvArgs ga{};
        ga.out = gates; ga.OCs = 896;
        ga.seg[0] = { cur,    c.wx, cur_bs,               Cin, 0,   7 };
        ga.seg[1] = { hb[li], c.wh, (long long)PIX * 64,  64,  448, 4 };
        ga.seg[2] = { mc_,    c.wm, (long long)PIX * 64,  64,  704, 3 };
        conv_wmma_kernel<3><<<dim3(256, 3), 256, smem3, stream>>>(ga);

        gates1_kernel<<<pwGrid, 256, 0, stream>>>(gates, c.bx, c.bh, c.bm,
                                                  cb[li], mc_, mn_, mem, oacc, totalG);

        // --- output convs over mem: wo (3x3, C=128) then wl (1x1, C=128) ---
        ConvArgs oa{};
        oa.out = ol; oa.OCs = 128;
        oa.seg[0] = { mem, c.wo, (long long)PIX * 128, 128, 0, 1 };
        conv_wmma_kernel<3><<<dim3(256, 1), 256, smem3, stream>>>(oa);

        ConvArgs la{};
        la.out = ol; la.OCs = 128;
        la.seg[0] = { mem, c.wl, (long long)PIX * 128, 128, 64, 1 };
        conv_wmma_kernel<1><<<dim3(256, 1), 256, smem1, stream>>>(la);

        gates2_kernel<<<pwGrid, 256, 0, stream>>>(oacc, ol, c.bo, c.bl, hb[li], totalG);

        float* tmp = mc_; mc_ = mn_; mn_ = tmp;
    };

    // encoder
    for (int t = 0; t < TSTEPS; ++t) {
        const float* cur = xnew + (long long)t * PIX * 9;
        long long cbs = (long long)TSTEPS * PIX * 9;
        int Cin = 9;
        for (int li = 0; li < 3; ++li) {
            cell_step(li, cur, cbs, Cin);
            cur = hb[li]; cbs = (long long)PIX * 64; Cin = 64;
        }
    }
    // decoder
    const float* dec = xnew + (long long)(TSTEPS - 1) * PIX * 9;
    for (int t = 0; t < HORIZON; ++t) {
        const float* cur = dec;
        long long cbs = (long long)TSTEPS * PIX * 9;
        int Cin = 9;
        for (int li = 0; li < 3; ++li) {
            cell_step(li, cur, cbs, Cin);
            cur = hb[li]; cbs = (long long)PIX * 64; Cin = 64;
        }
        int tot = BATCH * PIX;
        pred_kernel<<<(tot + 255) / 256, 256, 0, stream>>>(hb[2], P.w_last, P.b_last,
                                                           (float*)d_out, t);
    }
}
